// SelfAttention_57552561767180
// MI455X (gfx1250) — compile-verified, matrix operand order checked by reading, not run on hardware
//
#include <hip/hip_runtime.h>
#include <hip/hip_bf16.h>

// ---------------- problem constants ----------------
constexpr int S    = 4096;
constexpr int HID  = 2048;
constexpr int NH   = 16;
constexpr int NKV  = 4;
constexpr int HD   = 128;
constexpr int WIN  = 1024;
constexpr int QD   = NH * HD;    // 2048
constexpr int KVD  = NKV * HD;   // 512
constexpr float SCALE = 0.088388347648318447f;  // 1/sqrt(128)

typedef __bf16 bf16_t;
typedef __attribute__((ext_vector_type(8)))  __bf16 v8bf;
typedef __attribute__((ext_vector_type(16))) __bf16 v16bf;
typedef __attribute__((ext_vector_type(8)))  float  v8f;

#if __has_builtin(__builtin_amdgcn_global_load_async_to_lds_b128)
#define HAVE_ASYNC_LDS 1
#endif

// exact parameter types of the async-to-LDS builtin (from compiler diagnostic):
//   (v4i addrspace(1)*, v4i addrspace(3)*, imm offset, imm cpol)
typedef int v4i_vs __attribute__((vector_size(16)));
typedef __attribute__((address_space(1))) v4i_vs as1_v4i;
typedef __attribute__((address_space(3))) v4i_vs as3_v4i;

// 16-byte global -> LDS copy; async DMA (ASYNCcnt) when available.
static __device__ inline void cp16(const bf16_t* g, bf16_t* l) {
#ifdef HAVE_ASYNC_LDS
  __builtin_amdgcn_global_load_async_to_lds_b128((as1_v4i*)(void*)g,
                                                 (as3_v4i*)(void*)l, 0, 0);
#else
  *reinterpret_cast<v8bf*>(l) = *reinterpret_cast<const v8bf*>(g);
#endif
}

template <int N>
static __device__ inline void wait_async() {
#ifdef HAVE_ASYNC_LDS
#if __has_builtin(__builtin_amdgcn_s_wait_asynccnt)
  __builtin_amdgcn_s_wait_asynccnt(N);
#else
  if constexpr (N == 0)      asm volatile("s_wait_asynccnt 0x0" ::: "memory");
  else if constexpr (N == 4) asm volatile("s_wait_asynccnt 0x4" ::: "memory");
  else                       asm volatile("s_wait_asynccnt 0x6" ::: "memory");
#endif
#endif
}

static __device__ inline v16bf cat16(v8bf lo, v8bf hi) {
  return __builtin_shufflevector(lo, hi, 0,1,2,3,4,5,6,7,8,9,10,11,12,13,14,15);
}

// 16(row) x 32(K) bf16 WMMA fragment load (A layout; B mirrored with row=N).
// Lane L<16 : row=L,    K = {0..7, 16..23}
// Lane L>=16: row=L-16, K = {8..15, 24..31}
static __device__ inline v16bf load_frag(const bf16_t* base, int ld, int lane) {
  const int r   = lane & 15;
  const int off = (lane >> 4) << 3;            // 0 or 8
  const bf16_t* p = base + (size_t)r * ld + off;
  v8bf lo = *reinterpret_cast<const v8bf*>(p);
  v8bf hi = *reinterpret_cast<const v8bf*>(p + 16);
  return cat16(lo, hi);
}

static __device__ inline v8f wmma_bf16(v16bf a, v16bf b, v8f c) {
  return __builtin_amdgcn_wmma_f32_16x16x32_bf16(false, a, false, b,
                                                 (short)0, c, false, false);
}

// ---------------- kernel 0: f32 -> bf16 convert ----------------
__global__ __launch_bounds__(256) void cvt_f32_bf16(const float* __restrict__ in,
                                                    bf16_t* __restrict__ out, int n) {
  int i = blockIdx.x * 256 + threadIdx.x;
  if (i < n) out[i] = (bf16_t)in[i];
}

// ---------------- kernel 1: fused QKV GEMM + RoPE ----------------
// grid.x = S/64 (M tiles), grid.y = 24 N-blocks of 128:
//   0..15 -> q heads, 16..19 -> k heads, 20..23 -> v chunks
__global__ __launch_bounds__(256) void qkv_rope(
    const bf16_t* __restrict__ hid,   // [S, HID]
    const bf16_t* __restrict__ wqb,   // [QD,  HID]
    const bf16_t* __restrict__ wkb,   // [KVD, HID]
    const bf16_t* __restrict__ wvb,   // [KVD, HID]
    const float*  __restrict__ cosb,  // [S, HD]
    const float*  __restrict__ sinb,  // [S, HD]
    bf16_t* __restrict__ qout,        // [S, QD]
    bf16_t* __restrict__ kout,        // [S, KVD]
    bf16_t* __restrict__ vout)        // [S, KVD]
{
  __shared__ __align__(16) bf16_t As[2][64][72];    // 64 rows x 64 K, double buf
  __shared__ __align__(16) bf16_t Bs[2][128][72];   // 128 N  x 64 K, double buf
  __shared__ __align__(16) float  tile[64][132];

  const int mblk = blockIdx.x;
  const int nblk = blockIdx.y;
  const int tid  = threadIdx.x;
  const int wid  = tid >> 5, lane = tid & 31;
  const int wm = wid & 3, wn = wid >> 2;       // 4 waves along M, 2 along N
  const int ncol0 = wn * 64;
  const int rhi   = (lane >> 4) << 3;

  const bf16_t* wbase;
  bool rope;
  if (nblk < 16)      { wbase = wqb + (size_t)nblk * 128 * HID;        rope = true;  }
  else if (nblk < 20) { wbase = wkb + (size_t)(nblk - 16) * 128 * HID; rope = true;  }
  else                { wbase = wvb + (size_t)(nblk - 20) * 128 * HID; rope = false; }
  const bf16_t* abase = hid + (size_t)(mblk * 64) * HID;

  // stage one 64-wide K slab (A: 2 chunks/thread, B: 4 chunks/thread = 6 async ops/wave)
  auto stage = [&](int buf, int kt) {
#pragma unroll
    for (int j = 0; j < 2; ++j) {
      const int idx = tid + j * 256;            // 512 chunks
      const int r = idx >> 3, c = idx & 7;
      cp16(abase + (size_t)r * HID + kt * 64 + c * 8, &As[buf][r][c * 8]);
    }
#pragma unroll
    for (int j = 0; j < 4; ++j) {
      const int idx = tid + j * 256;            // 1024 chunks
      const int r = idx >> 3, c = idx & 7;
      cp16(wbase + (size_t)r * HID + kt * 64 + c * 8, &Bs[buf][r][c * 8]);
    }
  };

  v8f acc[4] = {};
  constexpr int NT = HID / 64;                  // 32 K-slabs
  stage(0, 0);
  for (int kt = 0; kt < NT; ++kt) {
    const int cur = kt & 1;
    if (kt + 1 < NT) { stage(cur ^ 1, kt + 1); wait_async<6>(); }
    else             { wait_async<0>(); }
    __syncthreads();
#pragma unroll
    for (int ks = 0; ks < 2; ++ks) {
      v16bf a = load_frag(&As[cur][wm * 16][ks * 32], 72, lane);
#pragma unroll
      for (int t = 0; t < 4; ++t) {
        v16bf b = load_frag(&Bs[cur][ncol0 + t * 16][ks * 32], 72, lane);
        acc[t] = wmma_bf16(a, b, acc[t]);
      }
    }
    __syncthreads();
  }

#pragma unroll
  for (int t = 0; t < 4; ++t)
#pragma unroll
    for (int r = 0; r < 8; ++r)
      tile[wm * 16 + r + rhi][ncol0 + t * 16 + (lane & 15)] = acc[t][r];
  __syncthreads();

  for (int i = tid; i < 64 * 128; i += 256) {
    const int r = i >> 7, c = i & 127;
    const int s = mblk * 64 + r;
    float x = tile[r][c];
    float y = x;
    if (rope) {
      // reference rope: tmp = concat(second_half, first_half) -> partner d^64
      float prt = tile[r][c ^ 64];
      y = x * cosb[s * HD + c] + prt * sinb[s * HD + c];
    }
    bf16_t o = (bf16_t)y;
    if (nblk < 16)      qout[(size_t)s * QD  + nblk * 128 + c]        = o;
    else if (nblk < 20) kout[(size_t)s * KVD + (nblk - 16) * 128 + c] = o;
    else                vout[(size_t)s * KVD + (nblk - 20) * 128 + c] = o;
  }
}

// ---------------- kernel 2: sliding-window flash attention ----------------
// grid.x = S/64 (query tiles), grid.y = NH (query heads)
__global__ __launch_bounds__(256) void flash_attn(
    const bf16_t* __restrict__ qb,   // [S, QD]
    const bf16_t* __restrict__ kb,   // [S, KVD]
    const bf16_t* __restrict__ vb,   // [S, KVD]
    bf16_t* __restrict__ ob)         // [S, QD]
{
  __shared__ __align__(16) bf16_t Ks[2][64][136];  // [key][d], double buffered
  __shared__ __align__(16) float  Sx[64][68];
  __shared__ __align__(16) bf16_t P[64][72];
  __shared__ __align__(16) bf16_t Vt[128][72];     // [d][key]
  __shared__ float rowm[64], rowl[64], rowf[64];
  __shared__ float pmax[64][4], psum[64][4];

  const int qblk = blockIdx.x;
  const int h    = blockIdx.y;
  const int hk   = h >> 2;          // GQA: 4 q heads per kv head
  const int qb0  = qblk * 64;
  const int tid  = threadIdx.x;
  const int wid  = tid >> 5, lane = tid & 31;
  const int wm = wid & 3, wn = wid >> 2;
  const int rhi = (lane >> 4) << 3;

  // async-stage one K tile: 64 keys x 128 d -> 1024 chunks, 4 per thread
  auto stageK = [&](int buf, int kbi) {
    const int k0 = kbi * 64;
#pragma unroll
    for (int j = 0; j < 4; ++j) {
      const int idx = tid + j * 256;
      const int row = idx >> 4, c = idx & 15;
      cp16(kb + (size_t)(k0 + row) * KVD + hk * HD + c * 8, &Ks[buf][row][c * 8]);
    }
  };

  // Q fragments for this wave's 16 rows, all 128 K
  v16bf qf[4];
#pragma unroll
  for (int kk = 0; kk < 4; ++kk)
    qf[kk] = load_frag(qb + (size_t)(qb0 + wm * 16) * QD + h * HD + kk * 32, QD, lane);

  v8f oacc[4] = {};

  if (tid < 64) { rowm[tid] = -1e30f; rowl[tid] = 0.f; }

  int kstart = qb0 - (WIN - 1); if (kstart < 0) kstart = 0;
  const int kb_lo = kstart >> 6;
  const int kb_hi = (qb0 + 63) >> 6;

  stageK(0, kb_lo);
  __syncthreads();

  for (int kbi = kb_lo; kbi <= kb_hi; ++kbi) {
    const int k0  = kbi * 64;
    const int cur = (kbi - kb_lo) & 1;

    // prefetch next K tile while we stage V and compute
    if (kbi + 1 <= kb_hi) stageK(cur ^ 1, kbi + 1);

    // stage V^T tile: Vt[d][key] (transpose -> element-wise)
    for (int i = tid; i < 64 * 128; i += 256) {
      const int key = i >> 7, d = i & 127;
      Vt[d][key] = vb[(size_t)(k0 + key) * KVD + hk * HD + d];
    }
    if (kbi + 1 <= kb_hi) wait_async<4>(); else wait_async<0>();
    __syncthreads();   // Ks[cur] + Vt visible

    // S = Q K^T for this wave's 16q x 32k sub-tile (K frags from LDS)
    v8f sacc[2] = {};
#pragma unroll
    for (int kk = 0; kk < 4; ++kk) {
#pragma unroll
      for (int t = 0; t < 2; ++t) {
        v16bf bfrag = load_frag(&Ks[cur][wn * 32 + t * 16][kk * 32], 136, lane);
        sacc[t] = wmma_bf16(qf[kk], bfrag, sacc[t]);
      }
    }
    // scale + causal sliding-window mask, stash in LDS
#pragma unroll
    for (int t = 0; t < 2; ++t)
#pragma unroll
      for (int r = 0; r < 8; ++r) {
        const int qr = qb0 + wm * 16 + r + rhi;
        const int kc = k0 + wn * 32 + t * 16 + (lane & 15);
        float sv = sacc[t][r] * SCALE;
        const int dd = qr - kc;
        if (dd < 0 || dd >= WIN) sv = -1e30f;
        Sx[wm * 16 + r + rhi][wn * 32 + t * 16 + (lane & 15)] = sv;
      }
    __syncthreads();

    // online softmax: 4 threads per row, 16 cols each
    {
      const int row = tid & 63, sub = tid >> 6;
      float mx = -1e30f;
#pragma unroll
      for (int c = 0; c < 16; ++c) mx = fmaxf(mx, Sx[row][sub * 16 + c]);
      pmax[row][sub] = mx;
    }
    __syncthreads();
    if (tid < 64) {
      float mx = fmaxf(fmaxf(pmax[tid][0], pmax[tid][1]),
                       fmaxf(pmax[tid][2], pmax[tid][3]));
      float mnew = fmaxf(rowm[tid], mx);
      rowf[tid] = __expf(rowm[tid] - mnew);
      rowm[tid] = mnew;
    }
    __syncthreads();
    {
      const int row = tid & 63, sub = tid >> 6;
      const float mnew = rowm[row];
      float ssum = 0.f;
#pragma unroll
      for (int c = 0; c < 16; ++c) {
        float sv = Sx[row][sub * 16 + c];
        float p  = __expf(sv - mnew);
        if (sv < -1e29f) p = 0.f;   // fully-masked row guard
        P[row][sub * 16 + c] = (bf16_t)p;
        ssum += p;
      }
      psum[row][sub] = ssum;
    }
    __syncthreads();
    if (tid < 64)
      rowl[tid] = rowl[tid] * rowf[tid]
                + psum[tid][0] + psum[tid][1] + psum[tid][2] + psum[tid][3];

    // rescale O accumulators by exp(m_old - m_new)
#pragma unroll
    for (int t = 0; t < 4; ++t)
#pragma unroll
      for (int r = 0; r < 8; ++r)
        oacc[t][r] *= rowf[wm * 16 + r + rhi];

    // O += P @ V  (A from P in LDS, B from Vt in LDS)
    v16bf pf[2];
#pragma unroll
    for (int j = 0; j < 2; ++j)
      pf[j] = load_frag(&P[wm * 16][j * 32], 72, lane);
#pragma unroll
    for (int t = 0; t < 4; ++t) {
#pragma unroll
      for (int j = 0; j < 2; ++j) {
        v16bf vfrag = load_frag(&Vt[wn * 64 + t * 16][j * 32], 72, lane);
        oacc[t] = wmma_bf16(pf[j], vfrag, oacc[t]);
      }
    }
    __syncthreads();
  }

  // normalize and store bf16 attention output
#pragma unroll
  for (int t = 0; t < 4; ++t)
#pragma unroll
    for (int r = 0; r < 8; ++r) {
      const int row = wm * 16 + r + rhi;
      const int qr  = qb0 + row;
      const int dc  = wn * 64 + t * 16 + (lane & 15);
      float o = oacc[t][r] / rowl[row];
      ob[(size_t)qr * QD + h * HD + dc] = (bf16_t)o;
    }
}

// ---------------- kernel 3: output projection ----------------
// grid.x = S/64, grid.y = HID/128
__global__ __launch_bounds__(256) void out_proj(
    const bf16_t* __restrict__ attn,  // [S, QD]
    const bf16_t* __restrict__ wob,   // [HID, QD]
    float* __restrict__ out)          // [S, HID]
{
  __shared__ __align__(16) bf16_t As[2][64][72];
  __shared__ __align__(16) bf16_t Bs[2][128][72];

  const int mblk = blockIdx.x;
  const int nblk = blockIdx.y;
  const int tid = threadIdx.x;
  const int wid = tid >> 5, lane = tid & 31;
  const int wm = wid & 3, wn = wid >> 2;
  const int ncol0 = wn * 64;
  const int rhi   = (lane >> 4) << 3;

  const bf16_t* abase = attn + (size_t)(mblk * 64) * QD;
  const bf16_t* wbase = wob + (size_t)(nblk * 128) * QD;

  auto stage = [&](int buf, int kt) {
#pragma unroll
    for (int j = 0; j < 2; ++j) {
      const int idx = tid + j * 256;
      const int r = idx >> 3, c = idx & 7;
      cp16(abase + (size_t)r * QD + kt * 64 + c * 8, &As[buf][r][c * 8]);
    }
#pragma unroll
    for (int j = 0; j < 4; ++j) {
      const int idx = tid + j * 256;
      const int r = idx >> 3, c = idx & 7;
      cp16(wbase + (size_t)r * QD + kt * 64 + c * 8, &Bs[buf][r][c * 8]);
    }
  };

  v8f acc[4] = {};
  constexpr int NT = QD / 64;
  stage(0, 0);
  for (int kt = 0; kt < NT; ++kt) {
    const int cur = kt & 1;
    if (kt + 1 < NT) { stage(cur ^ 1, kt + 1); wait_async<6>(); }
    else             { wait_async<0>(); }
    __syncthreads();
#pragma unroll
    for (int ks = 0; ks < 2; ++ks) {
      v16bf a = load_frag(&As[cur][wm * 16][ks * 32], 72, lane);
#pragma unroll
      for (int t = 0; t < 4; ++t) {
        v16bf b = load_frag(&Bs[cur][ncol0 + t * 16][ks * 32], 72, lane);
        acc[t] = wmma_bf16(a, b, acc[t]);
      }
    }
    __syncthreads();
  }

#pragma unroll
  for (int t = 0; t < 4; ++t)
#pragma unroll
    for (int r = 0; r < 8; ++r)
      out[(size_t)(mblk * 64 + wm * 16 + r + rhi) * HID
          + nblk * 128 + ncol0 + t * 16 + (lane & 15)] = acc[t][r];
}

// ---------------- launcher ----------------
extern "C" void kernel_launch(void* const* d_in, const int* in_sizes, int n_in,
                              void* d_out, int out_size, void* d_ws, size_t ws_size,
                              hipStream_t stream)
{
  (void)in_sizes; (void)n_in; (void)out_size; (void)ws_size;
  const float* hid  = (const float*)d_in[0];
  const float* wq   = (const float*)d_in[1];
  const float* wk   = (const float*)d_in[2];
  const float* wv   = (const float*)d_in[3];
  const float* wo   = (const float*)d_in[4];
  const float* cosb = (const float*)d_in[5];
  const float* sinb = (const float*)d_in[6];
  float* out = (float*)d_out;

  char* ws = (char*)d_ws;
  size_t off = 0;
  auto take = [&](size_t elems) -> bf16_t* {
    bf16_t* p = (bf16_t*)(ws + off);
    off += (elems * sizeof(bf16_t) + 255) & ~(size_t)255;
    return p;
  };
  bf16_t* hid_bf = take((size_t)S * HID);
  bf16_t* wq_bf  = take((size_t)QD * HID);
  bf16_t* wk_bf  = take((size_t)KVD * HID);
  bf16_t* wv_bf  = take((size_t)KVD * HID);
  bf16_t* wo_bf  = take((size_t)HID * QD);
  bf16_t* q_bf   = take((size_t)S * QD);
  bf16_t* k_bf   = take((size_t)S * KVD);
  bf16_t* v_bf   = take((size_t)S * KVD);
  bf16_t* a_bf   = take((size_t)S * QD);

  auto cvt = [&](const float* src, bf16_t* dst, int n) {
    cvt_f32_bf16<<<(n + 255) / 256, 256, 0, stream>>>(src, dst, n);
  };
  cvt(hid, hid_bf, S * HID);
  cvt(wq,  wq_bf,  QD * HID);
  cvt(wk,  wk_bf,  KVD * HID);
  cvt(wv,  wv_bf,  KVD * HID);
  cvt(wo,  wo_bf,  HID * QD);

  qkv_rope<<<dim3(S / 64, 24), 256, 0, stream>>>(hid_bf, wq_bf, wk_bf, wv_bf,
                                                 cosb, sinb, q_bf, k_bf, v_bf);
  flash_attn<<<dim3(S / 64, NH), 256, 0, stream>>>(q_bf, k_bf, v_bf, a_bf);
  out_proj<<<dim3(S / 64, HID / 128), 256, 0, stream>>>(a_bf, wo_bf, out);
}
